// RNN_816043786712
// MI455X (gfx1250) — matrix operand order repaired and checked by reading, not compiled, not run on hardware
//
#include <hip/hip_runtime.h>
#include <hip/hip_bf16.h>

// CDNA5 / gfx1250: wave32, WMMA 16x16x32 bf16 -> f32
typedef __attribute__((ext_vector_type(16))) __bf16 v16bf;
typedef __attribute__((ext_vector_type(8)))  float  v8f;

#define SEQ   512
#define BATCH 256
#define EMB   256
#define HID   512
#define OUTD  4
#define KTOT  (EMB + HID)      // 768 (concatenated K: embedding + hidden)
#define KT32  (KTOT / 32)      // 24 k-tiles
#define ASTRIDE (KTOT + 8)     // 776: pad so 16 rows hit distinct LDS banks
#define HPAD  (HID + 4)        // 516: pad f32 h_last rows
#define ROWS  16               // batch rows per workgroup

union FragBF { v16bf v; uint4 u[2]; };

// ---- prep: Wcat[n, 0:256] = W_ih[n,:], Wcat[n, 256:768] = W_hh[n,:]  (bf16)
//      bias[n] = b_ih[n] + b_hh[n]
__global__ void rnn_prep_kernel(const float* __restrict__ W_ih,
                                const float* __restrict__ W_hh,
                                const float* __restrict__ b_ih,
                                const float* __restrict__ b_hh,
                                __bf16* __restrict__ Wcat,
                                float* __restrict__ bias) {
    int gid = blockIdx.x * blockDim.x + threadIdx.x;
    const int total = HID * KTOT;
    if (gid < total) {
        int n = gid / KTOT;
        int k = gid - n * KTOT;
        float w = (k < EMB) ? W_ih[n * EMB + k] : W_hh[n * HID + (k - EMB)];
        Wcat[gid] = (__bf16)w;
    }
    if (gid < HID) bias[gid] = b_ih[gid] + b_hh[gid];
}

// ---- persistent recurrent kernel: 16 WGs x 256 threads, each WG owns 16
//      batch rows for all 512 timesteps. Hidden state stays in LDS (bf16).
__global__ __launch_bounds__(256)
void rnn_scan_kernel(const int*   __restrict__ text,   // [SEQ, BATCH]
                     const float* __restrict__ emb,    // [VOCAB, EMB]
                     const __bf16* __restrict__ Wcat,  // [HID, KTOT] bf16
                     const float* __restrict__ bias,   // [HID]
                     const float* __restrict__ fc_w,   // [OUTD, HID]
                     const float* __restrict__ fc_b,   // [OUTD]
                     float* __restrict__ out) {        // [BATCH, OUTD]
    __shared__ __align__(16) __bf16 Abuf[ROWS * ASTRIDE];  // [emb(256) | h(512)] per row
    __shared__ float hlast[ROWS * HPAD];                   // f32 h at s = SEQ-1

    const int tid  = threadIdx.x;
    const int lane = tid & 31;
    const int wave = tid >> 5;            // 0..7
    const int b0   = blockIdx.x * ROWS;   // first batch row of this WG

    // zero initial hidden state region
    for (int i = tid; i < ROWS * HID; i += 256) {
        int m = i / HID, k = i - m * HID;
        Abuf[m * ASTRIDE + EMB + k] = (__bf16)0.0f;
    }
    __syncthreads();

    const int frow   = lane & 15;          // A: M index / B,C: N index within tile
    const int ahalf  = (lane >> 4) * 8;    // A k-offset per lane half (ISA 16-bit A layout)
    const int bhalf  = (lane >> 4) * 8;    // B k-offset per lane half (ISA 16-bit B layout)
    const int rowoff = (lane >> 4) * 8;    // C: lanes 16-31 hold M = vgpr + 8

    // Per-lane base pointers. Every operand address below is base + constant.
    const __bf16* abase = &Abuf[frow * ASTRIDE + ahalf];                      // + kt*32 (+16)
    const __bf16* bbase = Wcat + (size_t)(wave * 64 + frow) * KTOT + bhalf;   // + nt*16*KTOT + kt*32

    // gather indexing (constant across steps)
    const int gm    = tid >> 4;            // 0..15 batch row
    const int gch   = (tid & 15) * 16;     // 16-float chunk
    __bf16* gdst    = &Abuf[gm * ASTRIDE + gch];

    v8f acc[4];
    const v8f vzero = {0.f, 0.f, 0.f, 0.f, 0.f, 0.f, 0.f, 0.f};

    for (int s = 0; s < SEQ; ++s) {
        // --- gather this step's embedding rows -> bf16 in LDS A[:, 0:EMB]
        {
            const int token = text[s * BATCH + b0 + gm];
            const float* src = emb + (long long)token * EMB + gch;
            #pragma unroll
            for (int j = 0; j < 16; ++j) gdst[j] = (__bf16)src[j];
            // prefetch next step's embedding rows into near caches
            if (s + 1 < SEQ) {
                const int tok2 = text[(s + 1) * BATCH + b0 + gm];
                __builtin_prefetch(emb + (long long)tok2 * EMB + gch, 0, 3);
            }
        }
        __syncthreads();

        // Launder a zero offset once per timestep. The volatile asm makes the
        // offset opaque, so LLVM cannot hoist/CSE the 96 weight-fragment loads
        // out of the s-loop (the hoist caused a 3KB/lane scratch spill before).
        // Unlike laundering the pointer itself, this keeps the address-space
        // provenance (kernel-arg -> global), so the loads lower to
        // global_load_b128 (LOADcnt only) instead of flat_load_b128
        // (LOADcnt+DScnt, aperture check coupled to the LDS pipe).
        int zoff = 0;
        asm volatile("" : "+v"(zoff));
        const __bf16* bptr = bbase + zoff;

        // --- GEMM: [16 x 768] x [768 x 512], this wave does 4 N-tiles of 16
        #pragma unroll
        for (int nt = 0; nt < 4; ++nt) acc[nt] = vzero;

        #pragma unroll
        for (int kt = 0; kt < KT32; ++kt) {
            FragBF a;
            a.u[0] = *reinterpret_cast<const uint4*>(abase + kt * 32);       // K0+ahalf..+7
            a.u[1] = *reinterpret_cast<const uint4*>(abase + kt * 32 + 16);  // K0+16+ahalf..+7
            #pragma unroll
            for (int nt = 0; nt < 4; ++nt) {
                const __bf16* bp = bptr + nt * (16 * KTOT) + kt * 32;        // K0+bhalf..+7, K0+16+bhalf..+7
                FragBF b;
                b.u[0] = *reinterpret_cast<const uint4*>(bp);
                b.u[1] = *reinterpret_cast<const uint4*>(bp + 16);
                acc[nt] = __builtin_amdgcn_wmma_f32_16x16x32_bf16(
                    false, a.v, false, b.v, (short)0, acc[nt], false, false);
            }
        }
        __syncthreads();   // all reads of Abuf (emb + h_prev) complete

        // --- epilogue: bias + tanh, write h_new back to LDS as bf16
        #pragma unroll
        for (int nt = 0; nt < 4; ++nt) {
            const int n  = (wave * 4 + nt) * 16 + frow;
            const float bv = bias[n];
            #pragma unroll
            for (int r = 0; r < 8; ++r) {
                const int m = r + rowoff;
                const float v = tanhf(acc[nt][r] + bv);
                Abuf[m * ASTRIDE + EMB + n] = (__bf16)v;
                if (s == SEQ - 1) hlast[m * HPAD + n] = v;
            }
        }
        __syncthreads();
    }

    // --- head: out[b, o] = h_last[b, :] . fc_w[o, :] + fc_b[o]
    if (tid < ROWS * OUTD) {
        const int m = tid >> 2, o = tid & 3;
        float sum = fc_b[o];
        const float* w = fc_w + o * HID;
        const float* h = &hlast[m * HPAD];
        for (int n = 0; n < HID; ++n) sum += h[n] * w[n];
        out[(b0 + m) * OUTD + o] = sum;
    }
}

extern "C" void kernel_launch(void* const* d_in, const int* in_sizes, int n_in,
                              void* d_out, int out_size, void* d_ws, size_t ws_size,
                              hipStream_t stream) {
    const int*   text = (const int*)  d_in[0];
    const float* emb  = (const float*)d_in[1];
    const float* W_ih = (const float*)d_in[2];
    const float* W_hh = (const float*)d_in[3];
    const float* b_ih = (const float*)d_in[4];
    const float* b_hh = (const float*)d_in[5];
    const float* fc_w = (const float*)d_in[6];
    const float* fc_b = (const float*)d_in[7];
    float* out = (float*)d_out;

    // workspace: Wcat bf16 [HID*KTOT] then bias f32 [HID]
    __bf16* Wcat = (__bf16*)d_ws;
    float*  bias = (float*)((char*)d_ws + (size_t)HID * KTOT * sizeof(__bf16));

    const int prepN = HID * KTOT;
    rnn_prep_kernel<<<(prepN + 255) / 256, 256, 0, stream>>>(
        W_ih, W_hh, b_ih, b_hh, Wcat, bias);

    rnn_scan_kernel<<<BATCH / ROWS, 256, 0, stream>>>(
        text, emb, Wcat, bias, fc_w, fc_b, out);
}
